// Block_45165876085372
// MI455X (gfx1250) — matrix operand order repaired and checked by reading, not compiled, not run on hardware
//
#include <hip/hip_runtime.h>
#include <hip/hip_bf16.h>
#include <stdint.h>

typedef __bf16 bf16;
typedef __attribute__((ext_vector_type(8)))  __bf16 v8bf;
typedef __attribute__((ext_vector_type(16))) __bf16 v16bf;
typedef __attribute__((ext_vector_type(8)))  float  v8f;

constexpr int Bc = 2, Sc = 1024, Pc = 1024, Dc = 1024, Hc = 16;
constexpr int DHc = Dc / Hc;          // 64
constexpr int Tc  = Pc + Sc;          // 2048
constexpr int BSc = Bc * Sc;          // 2048

// ---------------------------------------------------------- async-LDS helpers
// CDNA5: GLOBAL_LOAD_ASYNC_TO_LDS_B128 (GV mode), tracked by ASYNCcnt.
__device__ __forceinline__ void async_b128(void* lds, const void* g) {
  unsigned l = (unsigned)(uintptr_t)lds;            // generic LDS addr: low 32b = offset
  unsigned long long a = (unsigned long long)(uintptr_t)g;
  asm volatile("global_load_async_to_lds_b128 %0, %1, off" :: "v"(l), "v"(a) : "memory");
}
template <int N> __device__ __forceinline__ void wait_async() {
  asm volatile("s_wait_asynccnt %0" :: "n"(N) : "memory");
}

// ---------------------------------------------------------------- utilities
// transpose-cast: src fp32 [K,N] row-major -> dst bf16 [N,K]
__global__ __launch_bounds__(256) void cast_transpose_bf16(const float* __restrict__ src,
                                                           bf16* __restrict__ dst,
                                                           int K, int N) {
  int i = blockIdx.x * 256 + threadIdx.x;
  if (i >= K * N) return;
  int n = i % N, k = i / N;
  dst[(size_t)n * K + k] = (bf16)src[i];
}

// build bf16 concat caches: kcat [B,H,T,DH]; vcatT [B,H,DH,T] (t<P part)
__global__ __launch_bounds__(256) void cache_cast(const float* __restrict__ kc,
                                                  const float* __restrict__ vc,
                                                  bf16* __restrict__ kcat,
                                                  bf16* __restrict__ vcatT) {
  int i = blockIdx.x * 256 + threadIdx.x;       // over B*H*P*DH
  if (i >= Bc * Hc * Pc * DHc) return;
  int dh = i % DHc;
  int p  = (i / DHc) % Pc;
  int bh = i / (DHc * Pc);
  vcatT[((size_t)bh * DHc + dh) * Tc + p] = (bf16)vc[i];                       // v: [bh][p][dh]
  kcat[((size_t)bh * Tc + p) * DHc + dh] = (bf16)kc[((size_t)bh * DHc + dh) * Pc + p]; // k: [bh][dh][p]
}

__global__ __launch_bounds__(256) void layernorm_bf16(const float* __restrict__ x,
                                                      const float* __restrict__ w,
                                                      const float* __restrict__ b,
                                                      bf16* __restrict__ out) {
  int row = blockIdx.x;
  const float* xr = x + (size_t)row * Dc;
  __shared__ float red[256];
  float s = 0.f;
  for (int i = threadIdx.x; i < Dc; i += 256) s += xr[i];
  red[threadIdx.x] = s; __syncthreads();
  for (int st = 128; st > 0; st >>= 1) {
    if (threadIdx.x < st) red[threadIdx.x] += red[threadIdx.x + st];
    __syncthreads();
  }
  float mean = red[0] / Dc;
  __syncthreads();
  float v = 0.f;
  for (int i = threadIdx.x; i < Dc; i += 256) { float d = xr[i] - mean; v += d * d; }
  red[threadIdx.x] = v; __syncthreads();
  for (int st = 128; st > 0; st >>= 1) {
    if (threadIdx.x < st) red[threadIdx.x] += red[threadIdx.x + st];
    __syncthreads();
  }
  float rstd = rsqrtf(red[0] / Dc + 1e-12f);
  for (int i = threadIdx.x; i < Dc; i += 256)
    out[(size_t)row * Dc + i] = (bf16)(w[i] * ((xr[i] - mean) * rstd) + b[i]);
}

// scatter QKV: q->bf16 [B,H,S,DH]; new_k/new_v -> fp32 outputs + bf16 caches
__global__ __launch_bounds__(256) void qkv_scatter(const float* __restrict__ qkv,
                                                   bf16* __restrict__ qbf,
                                                   bf16* __restrict__ kcat,
                                                   bf16* __restrict__ vcatT,
                                                   float* __restrict__ nk_out,
                                                   float* __restrict__ nv_out) {
  int i = blockIdx.x * 256 + threadIdx.x;       // over B*S*D
  if (i >= Bc * Sc * Dc) return;
  int d = i % Dc;
  int s = (i / Dc) % Sc;
  int b = i / (Dc * Sc);
  int h = d / DHc, dh = d % DHc;
  size_t base = (size_t)(b * Sc + s) * (3 * Dc);
  float qv = qkv[base + d];
  float kv = qkv[base + Dc + d];
  float vv = qkv[base + 2 * Dc + d];
  int bh = b * Hc + h;
  qbf[((size_t)bh * Sc + s) * DHc + dh] = (bf16)qv;
  kcat[((size_t)bh * Tc + Pc + s) * DHc + dh] = (bf16)kv;
  vcatT[((size_t)bh * DHc + dh) * Tc + Pc + s] = (bf16)vv;
  nk_out[((size_t)bh * DHc + dh) * Sc + s] = kv;   // [B,H,DH,S]
  nv_out[((size_t)bh * Sc + s) * DHc + dh] = vv;   // [B,H,S,DH]
}

// ------------------------------------------------------- generic WMMA GEMM
// C[M,N] = A[M,K](bf16) @ Wt[N,K](bf16, pre-transposed) + bias, epilogue:
//   0: outF = acc+bias    1: outB = bf16(gelu(acc+bias))    2: outF = acc+bias+res
// Double-buffered GLOBAL_LOAD_ASYNC_TO_LDS_B128 staging (3 async instr / wave / stage).
__global__ __launch_bounds__(256) void gemm_bf16(const bf16* __restrict__ A,
                                                 const bf16* __restrict__ Wt,
                                                 const float* __restrict__ bias,
                                                 const float* __restrict__ res,
                                                 float* __restrict__ outF,
                                                 bf16* __restrict__ outB,
                                                 int M, int N, int K, int mode) {
  __shared__ bf16 sA[2][64][32];     // A tile, row-major      (2x4KB)
  __shared__ bf16 sW[2][128][32];    // W tile, [n][k]         (2x8KB)
  int tid = threadIdx.x;
  int lane = tid & 31, w = tid >> 5;
  int wr = w >> 1, wc = w & 1;
  int half = lane >> 4, lm = lane & 15;
  int m0 = blockIdx.y * 64;
  int n0 = blockIdx.x * 128;

  v8f acc[4] = {};

  auto stage = [&](int bsel, int k0) {
    { int r = tid >> 2, c = (tid & 3) * 8;          // 64 rows x 32
      async_b128(&sA[bsel][r][c], A + (size_t)(m0 + r) * K + k0 + c); }
#pragma unroll
    for (int it = 0; it < 2; ++it) {                // 128 rows x 32
      int idx = tid + it * 256;
      int n = idx >> 2, c = (idx & 3) * 8;
      async_b128(&sW[bsel][n][c], Wt + (size_t)(n0 + n) * K + k0 + c);
    }
  };

  int nk = K / 32;
  stage(0, 0);
  int buf = 0;
  for (int ks = 0; ks < nk; ++ks) {
    if (ks + 1 < nk) { stage(buf ^ 1, (ks + 1) * 32); wait_async<3>(); }
    else             { wait_async<0>(); }
    __syncthreads();

    v16bf af;
    {
      int m = wr * 16 + lm;
      int bk = half * 8;              // lanes 0-15: K {0..7,16..23}; 16-31: {8..15,24..31}
      v8bf lo = *(const v8bf*)(&sA[buf][m][bk]);
      v8bf hi = *(const v8bf*)(&sA[buf][m][bk + 16]);
#pragma unroll
      for (int i = 0; i < 8; ++i) { af[i] = lo[i]; af[i + 8] = hi[i]; }
    }
#pragma unroll
    for (int sub = 0; sub < 4; ++sub) {
      int n = wc * 64 + sub * 16 + lm;
      v8bf lo = *(const v8bf*)(&sW[buf][n][half * 8]);
      v8bf hi = *(const v8bf*)(&sW[buf][n][half * 8 + 16]);
      v16bf bfr;
#pragma unroll
      for (int i = 0; i < 8; ++i) { bfr[i] = lo[i]; bfr[i + 8] = hi[i]; }
      acc[sub] = __builtin_amdgcn_wmma_f32_16x16x32_bf16(
          false, af, false, bfr, (short)0, acc[sub], false, false);
    }
    __syncthreads();
    buf ^= 1;
  }

#pragma unroll
  for (int sub = 0; sub < 4; ++sub) {
    int gn = n0 + wc * 64 + sub * 16 + lm;
    float bv = bias[gn];
#pragma unroll
    for (int e = 0; e < 8; ++e) {
      int gm = m0 + wr * 16 + e + half * 8;
      size_t oi = (size_t)gm * N + gn;
      float vac = acc[sub][e] + bv;
      if (mode == 0) {
        outF[oi] = vac;
      } else if (mode == 1) {
        float t = 0.7978845608028654f * (vac + 0.044715f * vac * vac * vac);
        outB[oi] = (bf16)(0.5f * vac * (1.f + tanhf(t)));
      } else {
        outF[oi] = vac + res[oi];
      }
    }
  }
}

// ------------------------------------------------- flash attention (bf16 WMMA)
// grid: (B*H, S/128); block 256 (8 waves); each wave owns 16 query rows.
// K tile [t][dh] and V tile [dh][t] staged by async-to-LDS, double-buffered
// (2 async instr / wave / stage).
__global__ __launch_bounds__(256) void attn_kernel(const bf16* __restrict__ qbf,
                                                   const bf16* __restrict__ kcat,
                                                   const bf16* __restrict__ vcatT,
                                                   bf16* __restrict__ aout) {
  __shared__ bf16 sK[2][32][DHc];     // [t_local][dh]   (2x4KB)
  __shared__ bf16 sVt[2][DHc][32];    // [dh][t_local]   (2x4KB)
  __shared__ bf16 sP[8][16][32];      // per-wave probability tile (8KB)

  int tid = threadIdx.x, lane = tid & 31, w = tid >> 5;
  int half = lane >> 4, lm = lane & 15;
  int bh = blockIdx.x;
  int s0 = blockIdx.y * 128 + w * 16;

  v16bf aq[2];
  {
    const bf16* qrow = qbf + ((size_t)bh * Sc + s0 + lm) * DHc;
#pragma unroll
    for (int c = 0; c < 2; ++c) {
      int dh0 = c * 32 + half * 8;
      v8bf lo = *(const v8bf*)(qrow + dh0);
      v8bf hi = *(const v8bf*)(qrow + dh0 + 16);
#pragma unroll
      for (int i = 0; i < 8; ++i) { aq[c][i] = lo[i]; aq[c][i + 8] = hi[i]; }
    }
  }

  auto stage = [&](int bsel, int t0) {
    { int r = tid >> 3, c = (tid & 7) * 8;     // K: 32 rows x 64
      async_b128(&sK[bsel][r][c], kcat + ((size_t)bh * Tc + t0 + r) * DHc + c); }
    { int r = tid >> 2, c = (tid & 3) * 8;     // Vt: 64 rows x 32
      async_b128(&sVt[bsel][r][c], vcatT + ((size_t)bh * DHc + r) * Tc + t0 + c); }
  };

  float mrow[8], lrow[8];
  v8f oacc[4] = {};
#pragma unroll
  for (int e = 0; e < 8; ++e) { mrow[e] = -1e30f; lrow[e] = 0.f; }

  stage(0, 0);
  int buf = 0;
  for (int t0 = 0; t0 < Tc; t0 += 32) {
    if (t0 + 32 < Tc) { stage(buf ^ 1, t0 + 32); wait_async<2>(); }
    else              { wait_async<0>(); }
    __syncthreads();

    v8f sacc[2];
#pragma unroll
    for (int hh = 0; hh < 2; ++hh) {
      v8f a = {};
      int tl = hh * 16 + lm;
#pragma unroll
      for (int c = 0; c < 2; ++c) {
        const bf16* kp = &sK[buf][tl][c * 32 + half * 8];
        v8bf lo = *(const v8bf*)kp;
        v8bf hi = *(const v8bf*)(kp + 16);
        v16bf bfr;
#pragma unroll
        for (int i = 0; i < 8; ++i) { bfr[i] = lo[i]; bfr[i + 8] = hi[i]; }
        a = __builtin_amdgcn_wmma_f32_16x16x32_bf16(
            false, aq[c], false, bfr, (short)0, a, false, false);
      }
      sacc[hh] = a;
    }

    // online softmax (row reductions across the 16-lane half)
#pragma unroll
    for (int e = 0; e < 8; ++e) {
      float v0 = sacc[0][e], v1 = sacc[1][e];
      float tm = fmaxf(v0, v1);
      for (int msk = 8; msk >= 1; msk >>= 1) tm = fmaxf(tm, __shfl_xor(tm, msk, 16));
      float mnew = fmaxf(mrow[e], tm);
      float p0 = __expf(v0 - mnew), p1 = __expf(v1 - mnew);
      float ps = p0 + p1;
      for (int msk = 8; msk >= 1; msk >>= 1) ps += __shfl_xor(ps, msk, 16);
      float scale = __expf(mrow[e] - mnew);
      lrow[e] = lrow[e] * scale + ps;
      mrow[e] = mnew;
#pragma unroll
      for (int sub = 0; sub < 4; ++sub) oacc[sub][e] *= scale;
      sP[w][e + half * 8][lm]      = (bf16)p0;   // C-layout -> LDS
      sP[w][e + half * 8][16 + lm] = (bf16)p1;
    }

    // reload P as A-fragment (K = 32 keys)
    v16bf ap;
    {
      int bk = half * 8;
      const bf16* pp = &sP[w][lm][bk];
      v8bf lo = *(const v8bf*)pp;
      v8bf hi = *(const v8bf*)(pp + 16);
#pragma unroll
      for (int i = 0; i < 8; ++i) { ap[i] = lo[i]; ap[i + 8] = hi[i]; }
    }
#pragma unroll
    for (int sub = 0; sub < 4; ++sub) {
      const bf16* vp = &sVt[buf][sub * 16 + lm][half * 8];
      v8bf lo = *(const v8bf*)vp;
      v8bf hi = *(const v8bf*)(vp + 16);
      v16bf bfr;
#pragma unroll
      for (int i = 0; i < 8; ++i) { bfr[i] = lo[i]; bfr[i + 8] = hi[i]; }
      oacc[sub] = __builtin_amdgcn_wmma_f32_16x16x32_bf16(
          false, ap, false, bfr, (short)0, oacc[sub], false, false);
    }
    __syncthreads();
    buf ^= 1;
  }

  int b = bh / Hc, h = bh % Hc;
#pragma unroll
  for (int sub = 0; sub < 4; ++sub) {
    int dh = sub * 16 + lm;
#pragma unroll
    for (int e = 0; e < 8; ++e) {
      int s = s0 + e + half * 8;
      aout[((size_t)(b * Sc + s)) * Dc + h * DHc + dh] = (bf16)(oacc[sub][e] / lrow[e]);
    }
  }
}

// ------------------------------------------------------------------ driver
extern "C" void kernel_launch(void* const* d_in, const int* in_sizes, int n_in,
                              void* d_out, int out_size, void* d_ws, size_t ws_size,
                              hipStream_t stream) {
  const float* x      = (const float*)d_in[0];
  const float* kcache = (const float*)d_in[1];
  const float* vcache = (const float*)d_in[2];
  const float* ln1w   = (const float*)d_in[3];
  const float* ln1b   = (const float*)d_in[4];
  const float* ln2w   = (const float*)d_in[5];
  const float* ln2b   = (const float*)d_in[6];
  const float* w_attn = (const float*)d_in[7];
  const float* b_attn = (const float*)d_in[8];
  const float* w_proj = (const float*)d_in[9];
  const float* b_proj = (const float*)d_in[10];
  const float* w_fc   = (const float*)d_in[11];
  const float* b_fc   = (const float*)d_in[12];
  const float* w_fc2  = (const float*)d_in[13];
  const float* b_fc2  = (const float*)d_in[14];

  float* x_out  = (float*)d_out;                       // [B,S,D]
  float* nk_out = x_out + (size_t)BSc * Dc;            // [B,H,DH,S]
  float* nv_out = nk_out + (size_t)Bc * Hc * DHc * Sc; // [B,H,S,DH]

  char* ws = (char*)d_ws;
  size_t off = 0;
  auto alloc = [&](size_t bytes) { void* p = ws + off; off += (bytes + 255) & ~size_t(255); return p; };

  bf16* h1    = (bf16*)alloc((size_t)BSc * Dc * 2);
  bf16* wabT  = (bf16*)alloc((size_t)Dc * 3 * Dc * 2);      // [3D][D]
  bf16* wpbT  = (bf16*)alloc((size_t)Dc * Dc * 2);          // [D][D]
  bf16* wfbT  = (bf16*)alloc((size_t)Dc * 4 * Dc * 2);      // [4D][D]
  bf16* wf2bT = (bf16*)alloc((size_t)4 * Dc * Dc * 2);      // [D][4D]
  float* qkv  = (float*)alloc((size_t)BSc * 3 * Dc * 4);
  bf16* qbf   = (bf16*)alloc((size_t)Bc * Hc * Sc * DHc * 2);
  bf16* kcat  = (bf16*)alloc((size_t)Bc * Hc * Tc * DHc * 2);  // [B,H,T,DH]
  bf16* vcatT = (bf16*)alloc((size_t)Bc * Hc * DHc * Tc * 2);  // [B,H,DH,T]
  bf16* abf   = (bf16*)alloc((size_t)BSc * Dc * 2);
  float* x1   = (float*)alloc((size_t)BSc * Dc * 4);
  bf16* h2    = (bf16*)alloc((size_t)BSc * Dc * 2);
  bf16* mbf   = (bf16*)alloc((size_t)BSc * 4 * Dc * 2);

  auto cdiv = [](int a, int b) { return (a + b - 1) / b; };

  // weights -> bf16, transposed to [N][K]
  cast_transpose_bf16<<<cdiv(Dc * 3 * Dc, 256), 256, 0, stream>>>(w_attn, wabT, Dc, 3 * Dc);
  cast_transpose_bf16<<<cdiv(Dc * Dc, 256), 256, 0, stream>>>(w_proj, wpbT, Dc, Dc);
  cast_transpose_bf16<<<cdiv(Dc * 4 * Dc, 256), 256, 0, stream>>>(w_fc, wfbT, Dc, 4 * Dc);
  cast_transpose_bf16<<<cdiv(4 * Dc * Dc, 256), 256, 0, stream>>>(w_fc2, wf2bT, 4 * Dc, Dc);
  // KV caches -> bf16 concat buffers
  cache_cast<<<cdiv(Bc * Hc * Pc * DHc, 256), 256, 0, stream>>>(kcache, vcache, kcat, vcatT);

  // LN1
  layernorm_bf16<<<BSc, 256, 0, stream>>>(x, ln1w, ln1b, h1);
  // QKV = h1 @ w_attn + b_attn
  gemm_bf16<<<dim3(3 * Dc / 128, BSc / 64), 256, 0, stream>>>(
      h1, wabT, b_attn, nullptr, qkv, nullptr, BSc, 3 * Dc, Dc, 0);
  // scatter q / new_k / new_v
  qkv_scatter<<<cdiv(Bc * Sc * Dc, 256), 256, 0, stream>>>(qkv, qbf, kcat, vcatT, nk_out, nv_out);
  // attention
  attn_kernel<<<dim3(Bc * Hc, Sc / 128), 256, 0, stream>>>(qbf, kcat, vcatT, abf);
  // x1 = x + a @ w_proj + b_proj
  gemm_bf16<<<dim3(Dc / 128, BSc / 64), 256, 0, stream>>>(
      abf, wpbT, b_proj, x, x1, nullptr, BSc, Dc, Dc, 2);
  // LN2
  layernorm_bf16<<<BSc, 256, 0, stream>>>(x1, ln2w, ln2b, h2);
  // m = gelu(h2 @ w_fc + b_fc)
  gemm_bf16<<<dim3(4 * Dc / 128, BSc / 64), 256, 0, stream>>>(
      h2, wfbT, b_fc, nullptr, nullptr, mbf, BSc, 4 * Dc, Dc, 1);
  // x_out = x1 + m @ w_fc2 + b_fc2
  gemm_bf16<<<dim3(Dc / 128, BSc / 64), 256, 0, stream>>>(
      mbf, wf2bT, b_fc2, x1, x_out, nullptr, BSc, Dc, 4 * Dc, 2);

  (void)in_sizes; (void)n_in; (void)out_size; (void)ws_size;
}
